// FlashAttention_81252191306624
// MI455X (gfx1250) — matrix operand order repaired and checked by reading, not compiled, not run on hardware
//
#include <hip/hip_runtime.h>
#include <hip/hip_bf16.h>

typedef __attribute__((ext_vector_type(16))) __bf16        v16bf;
typedef __attribute__((ext_vector_type(8)))  float         v8f;
typedef __attribute__((ext_vector_type(4)))  unsigned int  u32x4;

#define DIM_   1024
#define SLEN_  2048
#define NH_    16
#define HD_    64
#define BATCH_ 4
#define MROWS_ 8192   /* B*S */

__device__ __forceinline__ v8f v8f_zero() {
    v8f z = {0.f, 0.f, 0.f, 0.f, 0.f, 0.f, 0.f, 0.f};
    return z;
}

// float -> bf16 bits, round-to-nearest-even
__device__ __forceinline__ unsigned short f2bf(float f) {
    unsigned int u = __builtin_bit_cast(unsigned int, f);
    if ((u & 0x7F800000u) == 0x7F800000u) return (unsigned short)(u >> 16); // inf/nan passthrough
    unsigned int r = u + 0x7FFFu + ((u >> 16) & 1u);
    return (unsigned short)(r >> 16);
}

union FragAB {            // one 16x32 (A) or 32x16 (B) bf16 fragment: 16 bf16 per lane
    u32x4 q[2];
    v16bf v;
};
union Pack8 {             // 8 bf16 packed for one 16B store
    unsigned short s[8];
    u32x4 q;
};

#define WMMA_BF16(A, B, C) \
    __builtin_amdgcn_wmma_f32_16x16x32_bf16(false, (A).v, false, (B).v, (short)0, (C), false, false)

// ---------------------------------------------------------------------------
// fp32 -> bf16 conversion (grid-stride)
// ---------------------------------------------------------------------------
__global__ __launch_bounds__(256)
void cvt_f32_bf16(const float* __restrict__ src, unsigned short* __restrict__ dst, int n) {
    int i = blockIdx.x * blockDim.x + threadIdx.x;
    int stride = gridDim.x * blockDim.x;
    for (; i < n; i += stride) dst[i] = f2bf(src[i]);
}

// ---------------------------------------------------------------------------
// GEMM helpers: fragment loads + one 8-WMMA k-step for a 32(M)x64(N) wave tile
// ---------------------------------------------------------------------------
__device__ __forceinline__ void load_a_frags(FragAB& a0, FragAB& a1,
                                             const unsigned short* arow0,
                                             const unsigned short* arow1,
                                             int kt, int g) {
    a0.q[0] = *(const u32x4*)(arow0 + kt + g * 8);
    a0.q[1] = *(const u32x4*)(arow0 + kt + 16 + g * 8);
    a1.q[0] = *(const u32x4*)(arow1 + kt + g * 8);
    a1.q[1] = *(const u32x4*)(arow1 + kt + 16 + g * 8);
}

__device__ __forceinline__ void load_b_frags(FragAB b[4], const unsigned short* W,
                                             int nBase, int ln, int kt, int g) {
#pragma unroll
    for (int fn = 0; fn < 4; ++fn) {
        const unsigned short* brow = W + (size_t)(nBase + fn * 16 + ln) * DIM_ + kt + g * 8;
        b[fn].q[0] = *(const u32x4*)(brow);
        b[fn].q[1] = *(const u32x4*)(brow + 16);
    }
}

__device__ __forceinline__ void mma_step(v8f acc[2][4], const FragAB& a0,
                                         const FragAB& a1, const FragAB b[4]) {
#pragma unroll
    for (int fn = 0; fn < 4; ++fn) {
        acc[0][fn] = WMMA_BF16(a0, b[fn], acc[0][fn]);
        acc[1][fn] = WMMA_BF16(a1, b[fn], acc[1][fn]);
    }
}

// ---------------------------------------------------------------------------
// C[m,n] = sum_k A[m,k] * W[n,k]   (A @ W.T), double-buffered k pipeline.
// Block 256 thr (8 waves) -> 128x128 tile; wave -> 32(M) x 64(N).
// MODE 0: bf16 row-major out.  MODE 1: bf16 out transposed as V^T [B][H][64][S].
// MODE 2: fp32 row-major out (final projection -> d_out).
// ---------------------------------------------------------------------------
template <int MODE>
__global__ __launch_bounds__(256)
void gemm_bf16_kernel(const unsigned short* __restrict__ A,
                      const unsigned short* __restrict__ W,
                      void* __restrict__ Cout) {
    const int tid  = threadIdx.x;
    const int wid  = tid >> 5;
    const int lane = tid & 31;
    const int g    = lane >> 4;
    const int ln   = lane & 15;

    const int mBase = blockIdx.x * 128 + (wid & 3) * 32;
    const int nBase = blockIdx.y * 128 + (wid >> 2) * 64;

    v8f acc[2][4];
#pragma unroll
    for (int i = 0; i < 2; ++i)
#pragma unroll
        for (int j = 0; j < 4; ++j) acc[i][j] = v8f_zero();

    const unsigned short* arow0 = A + (size_t)(mBase + ln) * DIM_;
    const unsigned short* arow1 = A + (size_t)(mBase + 16 + ln) * DIM_;

    // ping-pong fragment buffers: loads for step k+1 issue before WMMAs of step k
    FragAB a0A, a1A, a0B, a1B, bA[4], bB[4];
    load_a_frags(a0A, a1A, arow0, arow1, 0, g);
    load_b_frags(bA, W, nBase, ln, 0, g);

    for (int kt = 0; kt < DIM_; kt += 64) {
        load_a_frags(a0B, a1B, arow0, arow1, kt + 32, g);
        load_b_frags(bB, W, nBase, ln, kt + 32, g);
        if (kt + 128 < DIM_) {
            __builtin_prefetch(arow0 + kt + 128, 0, 1);
            __builtin_prefetch(W + (size_t)(nBase + lane) * DIM_ + kt + 128, 0, 1);
        }
        mma_step(acc, a0A, a1A, bA);

        if (kt + 64 < DIM_) {
            load_a_frags(a0A, a1A, arow0, arow1, kt + 64, g);
            load_b_frags(bA, W, nBase, ln, kt + 64, g);
        }
        mma_step(acc, a0B, a1B, bB);
    }

    if (MODE == 2) {
        float* C = (float*)Cout;
#pragma unroll
        for (int fm = 0; fm < 2; ++fm)
#pragma unroll
            for (int fn = 0; fn < 4; ++fn) {
                const int n  = nBase + fn * 16 + ln;
                const int mr = mBase + fm * 16 + g * 8;
#pragma unroll
                for (int r = 0; r < 8; ++r)
                    C[(size_t)(mr + r) * DIM_ + n] = acc[fm][fn][r];
            }
    } else if (MODE == 0) {
        unsigned short* C = (unsigned short*)Cout;
#pragma unroll
        for (int fm = 0; fm < 2; ++fm)
#pragma unroll
            for (int fn = 0; fn < 4; ++fn) {
                const int n  = nBase + fn * 16 + ln;
                const int mr = mBase + fm * 16 + g * 8;
#pragma unroll
                for (int r = 0; r < 8; ++r)
                    C[(size_t)(mr + r) * DIM_ + n] = f2bf(acc[fm][fn][r]);
            }
    } else {  // MODE 1: V^T  [B][H][64][S], 16B contiguous store per lane
        unsigned short* Vt = (unsigned short*)Cout;
#pragma unroll
        for (int fm = 0; fm < 2; ++fm)
#pragma unroll
            for (int fn = 0; fn < 4; ++fn) {
                const int n  = nBase + fn * 16 + ln;
                const int h  = n >> 6;
                const int d  = n & 63;
                const int mr = mBase + fm * 16 + g * 8;
                const int b  = mr >> 11;
                const int s  = mr & 2047;
                Pack8 p;
#pragma unroll
                for (int r = 0; r < 8; ++r) p.s[r] = f2bf(acc[fm][fn][r]);
                *(u32x4*)(Vt + ((((size_t)b * NH_ + h) * HD_ + d) << 11) + s) = p.q;
            }
    }
}

// ---------------------------------------------------------------------------
// Flash attention fragment loaders + one 32-key tile body
// ---------------------------------------------------------------------------
__device__ __forceinline__ void load_k_frags(FragAB bk[4], const unsigned short* krowbase,
                                             int kt, int g, int ln) {
#pragma unroll
    for (int ds = 0; ds < 2; ++ds) {
        const unsigned short* k0 = krowbase + (size_t)(kt + ln) * DIM_ + ds * 32 + g * 8;
        const unsigned short* k1 = krowbase + (size_t)(kt + 16 + ln) * DIM_ + ds * 32 + g * 8;
        bk[ds * 2 + 0].q[0] = *(const u32x4*)(k0);
        bk[ds * 2 + 0].q[1] = *(const u32x4*)(k0 + 16);
        bk[ds * 2 + 1].q[0] = *(const u32x4*)(k1);
        bk[ds * 2 + 1].q[1] = *(const u32x4*)(k1 + 16);
    }
}

__device__ __forceinline__ void load_v_frags(FragAB bv[4], const unsigned short* vbase,
                                             int kt, int g, int ln) {
#pragma unroll
    for (int df = 0; df < 4; ++df) {
        const unsigned short* vp = vbase + (size_t)(df * 16 + ln) * SLEN_ + kt + g * 8;
        bv[df].q[0] = *(const u32x4*)(vp);
        bv[df].q[1] = *(const u32x4*)(vp + 16);
    }
}

__device__ __forceinline__ void attn_tile(const FragAB aq[2], const FragAB bk[4],
                                          const FragAB bv[4], v8f o[4],
                                          float mrow[8], float lrow[8],
                                          unsigned short (*PldsW)[32], int g, int ln) {
    const float SCALE = 0.125f;  // 64^-0.5

    // ---- S = Q K^T: two 16-col D-fragments (K frags loaded one tile ago) ----
    v8f s0 = v8f_zero(), s1 = v8f_zero();
#pragma unroll
    for (int ds = 0; ds < 2; ++ds) {
        s0 = WMMA_BF16(aq[ds], bk[ds * 2 + 0], s0);
        s1 = WMMA_BF16(aq[ds], bk[ds * 2 + 1], s1);
    }

    // ---- online softmax (row stats across the owning 16-lane half) ---------
    float corr[8];
#pragma unroll
    for (int r = 0; r < 8; ++r) {
        float v0 = s0[r] * SCALE;
        float v1 = s1[r] * SCALE;
        float tm = fmaxf(v0, v1);
#pragma unroll
        for (int off = 8; off >= 1; off >>= 1)
            tm = fmaxf(tm, __shfl_xor(tm, off, 32));
        const float mnew = fmaxf(mrow[r], tm);
        const float p0 = __expf(v0 - mnew);
        const float p1 = __expf(v1 - mnew);
        float rs = p0 + p1;
#pragma unroll
        for (int off = 8; off >= 1; off >>= 1)
            rs += __shfl_xor(rs, off, 32);
        corr[r] = __expf(mrow[r] - mnew);
        lrow[r] = lrow[r] * corr[r] + rs;
        mrow[r] = mnew;
        PldsW[g * 8 + r][ln]      = f2bf(p0);
        PldsW[g * 8 + r][16 + ln] = f2bf(p1);
    }
#pragma unroll
    for (int df = 0; df < 4; ++df)
#pragma unroll
        for (int r = 0; r < 8; ++r) o[df][r] = o[df][r] * corr[r];

    __syncthreads();

    // ---- P D-frag -> A-frag via LDS, then O += P @ V (V already resident) ---
    FragAB ap;
    const u32x4* prow = (const u32x4*)(&PldsW[ln][0]);
    ap.q[0] = prow[g];
    ap.q[1] = prow[2 + g];

#pragma unroll
    for (int df = 0; df < 4; ++df) o[df] = WMMA_BF16(ap, bv[df], o[df]);

    __syncthreads();
}

// ---------------------------------------------------------------------------
// Flash attention. Grid (B*H, S/64), 128 threads (4 waves). Wave owns 16 q rows.
// Pipelined: V frags + next-tile K frags issue before S WMMAs; softmax covers
// the in-flight load latency; PV WMMAs consume resident V frags.
// ---------------------------------------------------------------------------
__global__ __launch_bounds__(128)
void flash_kernel(const unsigned short* __restrict__ Q,
                  const unsigned short* __restrict__ K,
                  const unsigned short* __restrict__ Vt,
                  unsigned short* __restrict__ O) {
    __shared__ __align__(16) unsigned short Plds[4][16][32];

    const int tid  = threadIdx.x;
    const int wid  = tid >> 5;
    const int lane = tid & 31;
    const int g    = lane >> 4;
    const int ln   = lane & 15;

    const int bh    = blockIdx.x;  // 0..63
    const int b     = bh >> 4;
    const int h     = bh & 15;
    const int qBase = blockIdx.y * 64 + wid * 16;

    // Q A-fragments (rows qBase+ln, d split 0..31 / 32..63) -- live whole kernel
    FragAB aq[2];
    {
        const unsigned short* qrow =
            Q + (size_t)(b * SLEN_ + qBase + ln) * DIM_ + h * HD_;
#pragma unroll
        for (int f = 0; f < 2; ++f) {
            aq[f].q[0] = *(const u32x4*)(qrow + f * 32 + g * 8);
            aq[f].q[1] = *(const u32x4*)(qrow + f * 32 + 16 + g * 8);
        }
    }

    v8f o[4];
#pragma unroll
    for (int df = 0; df < 4; ++df) o[df] = v8f_zero();
    float mrow[8], lrow[8];
#pragma unroll
    for (int r = 0; r < 8; ++r) { mrow[r] = -3.0e38f; lrow[r] = 0.f; }

    const unsigned short* krowbase = K + (size_t)b * SLEN_ * DIM_ + h * HD_;
    const unsigned short* vbase    = Vt + ((size_t)b * NH_ + h) * HD_ * SLEN_;

    FragAB bkA[4], bkB[4], bv[4];
    load_k_frags(bkA, krowbase, 0, g, ln);

    for (int kt = 0; kt < SLEN_; kt += 64) {
        // half 1: consume bkA (resident), stage bkB + V(kt)
        load_v_frags(bv, vbase, kt, g, ln);
        load_k_frags(bkB, krowbase, kt + 32, g, ln);
        if (kt + 64 < SLEN_)
            __builtin_prefetch(krowbase + (size_t)(kt + 64 + lane) * DIM_, 0, 1);
        attn_tile(aq, bkA, bv, o, mrow, lrow, Plds[wid], g, ln);

        // half 2: consume bkB (resident), stage bkA(kt+64) + V(kt+32)
        load_v_frags(bv, vbase, kt + 32, g, ln);
        if (kt + 64 < SLEN_)
            load_k_frags(bkA, krowbase, kt + 64, g, ln);
        attn_tile(aq, bkB, bv, o, mrow, lrow, Plds[wid], g, ln);
    }

    // ---- normalize + store bf16 [B*S, DIM] ----------------------------------
#pragma unroll
    for (int df = 0; df < 4; ++df) {
        const int d = h * HD_ + df * 16 + ln;
#pragma unroll
        for (int r = 0; r < 8; ++r) {
            const float v = o[df][r] / lrow[r];
            O[(size_t)(b * SLEN_ + qBase + g * 8 + r) * DIM_ + d] = f2bf(v);
        }
    }
}

// ---------------------------------------------------------------------------
extern "C" void kernel_launch(void* const* d_in, const int* in_sizes, int n_in,
                              void* d_out, int out_size, void* d_ws, size_t ws_size,
                              hipStream_t stream) {
    (void)in_sizes; (void)n_in; (void)out_size; (void)ws_size;
    const float* X  = (const float*)d_in[0];
    const float* Wq = (const float*)d_in[1];
    const float* Wk = (const float*)d_in[2];
    const float* Wv = (const float*)d_in[3];
    const float* Wo = (const float*)d_in[4];

    unsigned short* p   = (unsigned short*)d_ws;
    unsigned short* Xb  = p; p += (size_t)MROWS_ * DIM_;
    unsigned short* Wqb = p; p += (size_t)DIM_ * DIM_;
    unsigned short* Wkb = p; p += (size_t)DIM_ * DIM_;
    unsigned short* Wvb = p; p += (size_t)DIM_ * DIM_;
    unsigned short* Wob = p; p += (size_t)DIM_ * DIM_;
    unsigned short* Qb  = p; p += (size_t)MROWS_ * DIM_;
    unsigned short* Kb  = p; p += (size_t)MROWS_ * DIM_;
    unsigned short* Vtb = p; p += (size_t)MROWS_ * DIM_;
    unsigned short* Ob  = p; p += (size_t)MROWS_ * DIM_;

    cvt_f32_bf16<<<2048, 256, 0, stream>>>(X, Xb, MROWS_ * DIM_);
    cvt_f32_bf16<<<512, 256, 0, stream>>>(Wq, Wqb, DIM_ * DIM_);
    cvt_f32_bf16<<<512, 256, 0, stream>>>(Wk, Wkb, DIM_ * DIM_);
    cvt_f32_bf16<<<512, 256, 0, stream>>>(Wv, Wvb, DIM_ * DIM_);
    cvt_f32_bf16<<<512, 256, 0, stream>>>(Wo, Wob, DIM_ * DIM_);

    dim3 ggrid(MROWS_ / 128, DIM_ / 128);
    gemm_bf16_kernel<0><<<ggrid, 256, 0, stream>>>(Xb, Wqb, Qb);
    gemm_bf16_kernel<0><<<ggrid, 256, 0, stream>>>(Xb, Wkb, Kb);
    gemm_bf16_kernel<1><<<ggrid, 256, 0, stream>>>(Xb, Wvb, Vtb);

    flash_kernel<<<dim3(BATCH_ * NH_, SLEN_ / 64), 128, 0, stream>>>(Qb, Kb, Vtb, Ob);

    gemm_bf16_kernel<2><<<ggrid, 256, 0, stream>>>(Ob, Wob, d_out);
}
